// GCN_91173565759931
// MI455X (gfx1250) — matrix operand order repaired and checked by reading, not compiled, not run on hardware
//
#include <hip/hip_runtime.h>

#define D 128
#define BN_EPS 1e-5f

typedef __attribute__((ext_vector_type(16))) __bf16 v16bf;
typedef __attribute__((ext_vector_type(8)))  __bf16 v8bf;
typedef __attribute__((ext_vector_type(8)))  float  v8f;

// ---------------- zero fill (float4 granularity) ----------------
__global__ void fill_zero_kernel(float4* __restrict__ p, int n4) {
  int i = blockIdx.x * blockDim.x + threadIdx.x;
  if (i < n4) p[i] = make_float4(0.f, 0.f, 0.f, 0.f);
}

// ---------------- SpMM: agg[row] += val * x[col] ----------------
// 32 lanes per edge, float4 per lane: b128 gather, 4 f32 atomic scatter-adds.
__global__ void spmm_kernel(const int* __restrict__ rows, const int* __restrict__ cols,
                            const float* __restrict__ vals, const float* __restrict__ x,
                            float* __restrict__ agg, int nnz) {
  int idx = blockIdx.x * blockDim.x + threadIdx.x;
  int e = idx >> 5;
  if (e >= nnz) return;
  int q = (idx & 31) * 4;   // feature base
  int r = rows[e];
  int c = cols[e];
  float v = vals[e];
  float4 xv = *(const float4*)(x + (size_t)c * D + q);
  float* dst = agg + (size_t)r * D + q;
  atomicAdd(dst + 0, v * xv.x);
  atomicAdd(dst + 1, v * xv.y);
  atomicAdd(dst + 2, v * xv.z);
  atomicAdd(dst + 3, v * xv.w);
}

// ---------------- f32 -> bf16 (round to nearest even) ----------------
__global__ void cvt_bf16_kernel(const float* __restrict__ in, unsigned short* __restrict__ out,
                                int n) {
  int i = blockIdx.x * blockDim.x + threadIdx.x;
  if (i >= n) return;
  unsigned int u = __float_as_uint(in[i]);
  out[i] = (unsigned short)((u + 0x7FFFu + ((u >> 16) & 1u)) >> 16);
}

// ---------------- WMMA GEMM: out[m, n] = sum_k A[m,k] * W[n,k] ----------------
// Block = 256 threads (8 waves), 64 rows x 128 cols per block.
// A panel staged into LDS with gfx1250 async global->LDS copies (ASYNCcnt),
// then per-wave WMMA: 4 M-subtiles x 4 K-chunks = 16 v_wmma per wave.
__global__ __launch_bounds__(256)
void gemm_wmma_kernel(const unsigned short* __restrict__ Abf,
                      const unsigned short* __restrict__ Wbf,
                      float* __restrict__ out) {
  __shared__ __align__(16) unsigned short As[64 * D];  // 16 KB bf16 panel @ LDS offset 0

  const int tid = threadIdx.x;
  const int m0 = blockIdx.x * 64;

  // async panel load: 64*128*2B = 16KB = 4 passes * 256 lanes * 16B
  // GLOBAL_LOAD_ASYNC_TO_LDS_B128: VDST = LDS byte offset, VADDR = 64-bit addr (GV mode)
  const uint4* gsrc = (const uint4*)(Abf + (size_t)m0 * D);
#pragma unroll
  for (int p = 0; p < 4; ++p) {
    unsigned lds_off = (unsigned)((p * 256 + tid) * 16);
    unsigned long long ga = (unsigned long long)(gsrc + p * 256 + tid);
    asm volatile("global_load_async_to_lds_b128 %0, %1, off"
                 :: "v"(lds_off), "v"(ga) : "memory");
  }
  // Wait for this wave's async copies. Passing As as a (unused) VGPR operand
  // escapes the LDS object, so the "memory" clobbers above/here are assumed to
  // write it: the ds_loads below get a real def and cannot fold to undef.
  asm volatile("s_wait_asynccnt 0x0" :: "v"(As) : "memory");
  __syncthreads();

  const int wave = tid >> 5;   // N-tile index 0..7
  const int lane = tid & 31;
  const int hi   = lane >> 4;  // lane half selects K sub-range
  const int l16  = lane & 15;
  const int col  = wave * 16 + l16;

  // B fragments: B[k][n] = W[n][k] -> lane (col n) reads from W row n.
  // Lanes 0-15: K {0..7,16..23}, lanes 16-31: K {8..15,24..31} (matches A).
  const __bf16* Wb = (const __bf16*)Wbf;
  v16bf bfrag[4];
#pragma unroll
  for (int kc = 0; kc < 4; ++kc) {
    const __bf16* bp = Wb + (size_t)col * D + kc * 32 + hi * 8;
    union { v16bf v; v8bf h[2]; } bb;
    bb.h[0] = *(const v8bf*)(bp);
    bb.h[1] = *(const v8bf*)(bp + 16);
    bfrag[kc] = bb.v;
  }

  const __bf16* Asb = (const __bf16*)As;
#pragma unroll
  for (int ms = 0; ms < 4; ++ms) {
    v8f c = {0.f, 0.f, 0.f, 0.f, 0.f, 0.f, 0.f, 0.f};
#pragma unroll
    for (int kc = 0; kc < 4; ++kc) {
      // A fragment (ISA 16-bit A 16x32 layout): lane<16 -> K {0..7,16..23},
      // lane>=16 -> K {8..15,24..31}: two 16B LDS reads per lane.
      const __bf16* ap = Asb + (ms * 16 + l16) * D + kc * 32 + hi * 8;
      union { v16bf v; v8bf h[2]; } a;
      a.h[0] = *(const v8bf*)(ap);
      a.h[1] = *(const v8bf*)(ap + 16);
      c = __builtin_amdgcn_wmma_f32_16x16x32_bf16(
          /*neg_a=*/false, a.v, /*neg_b=*/false, bfrag[kc],
          /*c_mod=*/(short)0, c, /*reuse_a=*/false, /*reuse_b=*/false);
    }
    // C/D layout: VGPR v -> row (v + 8*hi), lane&15 -> col. Coalesced stores.
    float* orow = out + (size_t)(m0 + ms * 16 + hi * 8) * D + col;
#pragma unroll
    for (int v = 0; v < 8; ++v) orow[(size_t)v * D] = c[v];
  }
}

// ---------------- per-column sum / sum-of-squares for BN ----------------
__global__ void stats_kernel(const float* __restrict__ h, float* __restrict__ stats, int n) {
  int t = blockIdx.x * blockDim.x + threadIdx.x;
  int c = t & (D - 1);
  int rstride = (gridDim.x * blockDim.x) >> 7;
  float s = 0.f, sq = 0.f;
  for (int r = t >> 7; r < n; r += rstride) {
    float v = h[(size_t)r * D + c];
    s += v;
    sq += v * v;
  }
  atomicAdd(stats + c, s);
  atomicAdd(stats + D + c, sq);
}

// ---------------- BN (train-mode batch stats) + ReLU, in place ----------------
__global__ void bn_relu_kernel(float* __restrict__ h, const float* __restrict__ stats,
                               const float* __restrict__ g, const float* __restrict__ b, int n) {
  int idx = blockIdx.x * blockDim.x + threadIdx.x;
  if (idx >= n * D) return;
  int c = idx & (D - 1);
  float inv_n = 1.0f / (float)n;
  float mean = stats[c] * inv_n;
  float var = stats[D + c] * inv_n - mean * mean;
  float v = (h[idx] - mean) * rsqrtf(var + BN_EPS);
  float o = g[c] * v + b[c];
  h[idx] = o > 0.f ? o : 0.f;
}

// ---------------- head GEMV: y[row] = h[row,:] . Wout + bout ----------------
__global__ void gemv_kernel(const float* __restrict__ h, const float* __restrict__ Wo,
                            const float* __restrict__ bo, float* __restrict__ y, int n) {
  int row = blockIdx.x * (blockDim.x >> 5) + (threadIdx.x >> 5);
  int lane = threadIdx.x & 31;
  if (row >= n) return;
  float4 hv = *(const float4*)(h + (size_t)row * D + lane * 4);
  float4 wv = *(const float4*)(Wo + lane * 4);
  float s = hv.x * wv.x + hv.y * wv.y + hv.z * wv.z + hv.w * wv.w;
#pragma unroll
  for (int off = 16; off > 0; off >>= 1) s += __shfl_xor(s, off, 32);
  if (lane == 0) y[row] = s + bo[0];
}

extern "C" void kernel_launch(void* const* d_in, const int* in_sizes, int n_in,
                              void* d_out, int out_size, void* d_ws, size_t ws_size,
                              hipStream_t stream) {
  // setup_inputs order: x, A_indices, A_values, n_nodes, W1,g1,b1, W2,g2,b2, Wout,bout
  const float* x    = (const float*)d_in[0];
  const int*   Aidx = (const int*)d_in[1];
  const float* Aval = (const float*)d_in[2];
  const float* W1   = (const float*)d_in[4];
  const float* g1   = (const float*)d_in[5];
  const float* b1   = (const float*)d_in[6];
  const float* W2   = (const float*)d_in[7];
  const float* g2   = (const float*)d_in[8];
  const float* b2   = (const float*)d_in[9];
  const float* Wout = (const float*)d_in[10];
  const float* bout = (const float*)d_in[11];
  float* y = (float*)d_out;

  const int N    = in_sizes[0] / D;   // 50000 nodes
  const int NNZ  = in_sizes[2];       // 800000 edges
  const int NPAD = (N + 63) & ~63;    // pad to WMMA block M=64

  const int* rows = Aidx;            // A_indices[0]
  const int* cols = Aidx + NNZ;      // A_indices[1]

  // workspace carve-out
  char* w = (char*)d_ws;
  float* agg = (float*)w;                   w += (size_t)NPAD * D * sizeof(float);
  float* h   = (float*)w;                   w += (size_t)NPAD * D * sizeof(float);
  unsigned short* abf = (unsigned short*)w; w += (size_t)NPAD * D * sizeof(unsigned short);
  unsigned short* wbf = (unsigned short*)w; w += (size_t)D * D * sizeof(unsigned short);
  float* stats = (float*)w;                 w += 256 * sizeof(float);
  (void)ws_size; (void)n_in; (void)out_size;

  const int elemsPad = NPAD * D;
  dim3 b256(256);

  const float* hin = x;
  for (int layer = 0; layer < 2; ++layer) {
    const float* Wl = layer ? W2 : W1;
    const float* gl = layer ? g2 : g1;
    const float* bl = layer ? b2 : b1;

    fill_zero_kernel<<<(elemsPad / 4 + 255) / 256, b256, 0, stream>>>((float4*)agg, elemsPad / 4);
    spmm_kernel<<<(NNZ * 32 + 255) / 256, b256, 0, stream>>>(rows, cols, Aval, hin, agg, NNZ);
    cvt_bf16_kernel<<<(elemsPad + 255) / 256, b256, 0, stream>>>(agg, abf, elemsPad);
    cvt_bf16_kernel<<<(D * D + 255) / 256, b256, 0, stream>>>(Wl, wbf, D * D);
    gemm_wmma_kernel<<<NPAD / 64, b256, 0, stream>>>(abf, wbf, h);
    fill_zero_kernel<<<1, 64, 0, stream>>>((float4*)stats, 64);
    stats_kernel<<<512, b256, 0, stream>>>(h, stats, N);
    bn_relu_kernel<<<(N * D + 255) / 256, b256, 0, stream>>>(h, stats, gl, bl, N);
    hin = h;
  }
  gemv_kernel<<<(N + 7) / 8, b256, 0, stream>>>(h, Wout, bout, y, N);
}
